// GCGRU_27693949125277
// MI455X (gfx1250) — matrix operand order repaired and checked by reading, not compiled, not run on hardware
//
#include <hip/hip_runtime.h>
#include <math.h>

// ---------------------------------------------------------------------------
// GC-GRU (DCRNN-style) for MI455X / gfx1250, wave32 + WMMA bf16.
// N=1024 nodes, K=3 supports (k=0 identity -> skipped), H=64, C=1, B=64,
// T=12 encoder steps, HOR=12 decoder steps.
// Graph GEMM uses 32x64 per-wave tiles (2 A-frags x 4 B-frags, 8 accums)
// to maximize WMMA/byte; dense fused GEMM (K=224) uses 16x64 tiles.
// ---------------------------------------------------------------------------

typedef __attribute__((ext_vector_type(16))) __bf16 v16bf;
typedef __attribute__((ext_vector_type(8)))  float  v8f;

#define NN      1024
#define BB      64
#define HH      64
#define DIN     65            // C + H
#define QTOT    (BB * DIN)    // 4160 columns of the graph GEMM
#define KPAD    224           // 3*65 = 195 padded to 7*32
#define ROWS    (BB * NN)     // 65536 (b,i) rows

__device__ inline unsigned short f2bf(float f) {
    union { float f; unsigned u; } x; x.f = f;
    unsigned r = x.u + 0x7FFFu + ((x.u >> 16) & 1u);   // round-to-nearest-even
    return (unsigned short)(r >> 16);
}

// A fragment: 16x32 bf16, row-major source with leading dim `ld` (elements).
// lane m=lane&15 holds row m; chunks of 8 at K = 8*half and 16+8*half.
__device__ inline v16bf load_frag_A(const unsigned short* __restrict__ base,
                                    int ld, int row0, int kk, int lane) {
    int m = lane & 15, hf = lane >> 4;
    const unsigned short* p = base + (size_t)(row0 + m) * ld + kk + 8 * hf;
    union { v16bf v; uint4 u[2]; } t;
    t.u[0] = *(const uint4*)(p);
    t.u[1] = *(const uint4*)(p + 16);
    return t.v;
}

// B fragment: 32x16 bf16, source is B^T row-major ([col][K], leading dim ld).
// lane n=lane&15 holds column n; chunks of 8 at K = 8*group and 16+8*group.
__device__ inline v16bf load_frag_B(const unsigned short* __restrict__ base,
                                    int ld, int col0, int kk, int lane) {
    int n = lane & 15, g = lane >> 4;
    const unsigned short* p = base + (size_t)(col0 + n) * ld + kk + 8 * g;
    union { v16bf v; uint4 u[2]; } t;
    t.u[0] = *(const uint4*)(p);
    t.u[1] = *(const uint4*)(p + 16);
    return t.v;
}

#define WMMA_BF16(A, Bf, Cacc) \
    __builtin_amdgcn_wmma_f32_16x16x32_bf16(false, (A), false, (Bf), (short)0, (Cacc), false, false)

// -------------------------------- init kernels -----------------------------
__global__ void gcgru_zero_f32(float* __restrict__ p, int n) {
    int i = blockIdx.x * blockDim.x + threadIdx.x;
    if (i < n) p[i] = 0.0f;
}

// zero padding columns 192..223 of A2 (cols 192..194 are overwritten each step)
__global__ void gcgru_pad_a2(unsigned short* __restrict__ A2) {
    int r = blockIdx.x * blockDim.x + threadIdx.x;
    if (r >= ROWS) return;
    uint4 z = {0u, 0u, 0u, 0u};
    uint4* p = (uint4*)(A2 + (size_t)r * KPAD + 192);
    p[0] = z; p[1] = z; p[2] = z; p[3] = z;   // 32 bf16 = 64 B
}

// convert G[1] and G[2] (skip identity G[0]) to bf16 row-major
__global__ void gcgru_conv_g(const float* __restrict__ G, unsigned short* __restrict__ Gb) {
    int i = blockIdx.x * blockDim.x + threadIdx.x;
    if (i < 2 * NN * NN) Gb[i] = f2bf(G[NN * NN + i]);
}

// W: [195, OD] fp32  ->  WT: [OD, 224] bf16 (transposed, zero-padded K)
__global__ void gcgru_prep_w(const float* __restrict__ W, int OD,
                             unsigned short* __restrict__ WT) {
    int i = blockIdx.x * blockDim.x + threadIdx.x;
    if (i >= OD * KPAD) return;
    int o = i / KPAD, kc = i % KPAD;
    WT[i] = (kc < 3 * DIN) ? f2bf(W[kc * OD + o]) : (unsigned short)0;
}

// ----------------------------- build comb ----------------------------------
// Writes XT[q][j] (bf16, q=b*65+c, ld=1024) and A2[b*1024+j][c] (k=0 block).
// use_r=1 multiplies h by the reset gate r = zr[..., 64+c-1].
__global__ void gcgru_build(const float* __restrict__ xp, int xstride,
                            const float* __restrict__ h,
                            const float* __restrict__ zr, int use_r,
                            unsigned short* __restrict__ XT,
                            unsigned short* __restrict__ A2) {
    int q = blockIdx.x;            // 0..4159
    int b = q / DIN, c = q % DIN;
    for (int j = threadIdx.x; j < NN; j += blockDim.x) {
        float v;
        if (c == 0) {
            v = xp[(size_t)b * xstride + j];
        } else {
            size_t r = (size_t)(b * NN + j);
            v = h[r * HH + (c - 1)];
            if (use_r) v *= zr[r * 128 + HH + (c - 1)];
        }
        unsigned short bv = f2bf(v);
        XT[(size_t)q * NN + j] = bv;
        A2[(size_t)(b * NN + j) * KPAD + c] = bv;
    }
}

// ------------------------- graph GEMM: S_k = G_k @ X -----------------------
// Wave tile 32(M:i) x 64(N:q), K=1024. 8 accumulators, 2 A-frags reused over
// 4 B-frags each. Scatter-stores bf16 into the A2 k-block.
__global__ void gcgru_gemm_graph(const unsigned short* __restrict__ Gb,
                                 const unsigned short* __restrict__ XT,
                                 unsigned short* __restrict__ A2) {
    const int lane = threadIdx.x & 31;
    const int wave = threadIdx.x >> 5;
    const int kidx = blockIdx.z;                        // 0,1 -> support k=1,2
    const int i0 = blockIdx.y * 128 + wave * 32;
    const int q0 = blockIdx.x * 64;
    const unsigned short* Gk = Gb + (size_t)kidx * NN * NN;

    v8f c00 = {}, c01 = {}, c02 = {}, c03 = {};
    v8f c10 = {}, c11 = {}, c12 = {}, c13 = {};
    for (int kk = 0; kk < NN; kk += 32) {
        v16bf a0 = load_frag_A(Gk, NN, i0,      kk, lane);
        v16bf a1 = load_frag_A(Gk, NN, i0 + 16, kk, lane);
        v16bf b0 = load_frag_B(XT, NN, q0 +  0, kk, lane);
        v16bf b1 = load_frag_B(XT, NN, q0 + 16, kk, lane);
        v16bf b2 = load_frag_B(XT, NN, q0 + 32, kk, lane);
        v16bf b3 = load_frag_B(XT, NN, q0 + 48, kk, lane);
        c00 = WMMA_BF16(a0, b0, c00);
        c01 = WMMA_BF16(a0, b1, c01);
        c02 = WMMA_BF16(a0, b2, c02);
        c03 = WMMA_BF16(a0, b3, c03);
        c10 = WMMA_BF16(a1, b0, c10);
        c11 = WMMA_BF16(a1, b1, c11);
        c12 = WMMA_BF16(a1, b2, c12);
        c13 = WMMA_BF16(a1, b3, c13);
    }

    const int n = lane & 15, g = lane >> 4;
    const int kcol = DIN * (kidx + 1);
    auto scatter = [&](v8f acc, int mi, int f) {
        int q = q0 + f * 16 + n;
        int b = q / DIN, c = q % DIN;
        unsigned short* dst =
            A2 + (size_t)(b * NN + i0 + mi * 16 + 8 * g) * KPAD + kcol + c;
        #pragma unroll
        for (int v = 0; v < 8; ++v) dst[(size_t)v * KPAD] = f2bf(acc[v]);
    };
    scatter(c00, 0, 0); scatter(c01, 0, 1); scatter(c02, 0, 2); scatter(c03, 0, 3);
    scatter(c10, 1, 0); scatter(c11, 1, 1); scatter(c12, 1, 2); scatter(c13, 1, 3);
}

// ---------------- fused dense GEMM [ROWS x 224] @ [224 x OD] ---------------
// mode 0: gates = sigmoid(.) -> zr[ROWS x 128]
// mode 1: n = tanh(.);  h = z*n + (1-z)*h   (OD = 64, o0 = 0)
__global__ void gcgru_gemm_fuse(const unsigned short* __restrict__ A2,
                                const unsigned short* __restrict__ WT,
                                const float* __restrict__ bias, int mode,
                                float* __restrict__ zr, float* __restrict__ h) {
    const int lane = threadIdx.x & 31;
    const int wave = threadIdx.x >> 5;
    const int r0 = blockIdx.y * 64 + wave * 16;         // row = b*1024 + i
    const int o0 = blockIdx.x * 64;

    v8f acc0 = {}, acc1 = {}, acc2 = {}, acc3 = {};
    for (int kk = 0; kk < KPAD; kk += 32) {
        v16bf a  = load_frag_A(A2, KPAD, r0, kk, lane);
        v16bf b0 = load_frag_B(WT, KPAD, o0 +  0, kk, lane);
        v16bf b1 = load_frag_B(WT, KPAD, o0 + 16, kk, lane);
        v16bf b2 = load_frag_B(WT, KPAD, o0 + 32, kk, lane);
        v16bf b3 = load_frag_B(WT, KPAD, o0 + 48, kk, lane);
        acc0 = WMMA_BF16(a, b0, acc0);
        acc1 = WMMA_BF16(a, b1, acc1);
        acc2 = WMMA_BF16(a, b2, acc2);
        acc3 = WMMA_BF16(a, b3, acc3);
    }

    const int n = lane & 15, g = lane >> 4;
    auto epilogue = [&](v8f acc, int f) {
        int o = o0 + f * 16 + n;
        float bs = bias[o];
        #pragma unroll
        for (int v = 0; v < 8; ++v) {
            int r = r0 + 8 * g + v;
            float x = acc[v] + bs;
            if (mode == 0) {
                zr[(size_t)r * 128 + o] = 1.0f / (1.0f + __expf(-x));
            } else {
                float nv = tanhf(x);
                float z  = zr[(size_t)r * 128 + o];      // update gate (cols 0..63)
                size_t hi = (size_t)r * HH + o;
                float ho = h[hi];
                h[hi] = z * nv + (1.0f - z) * ho;
            }
        }
    };
    epilogue(acc0, 0); epilogue(acc1, 1); epilogue(acc2, 2); epilogue(acc3, 3);
}

// ------------------------------ projection ---------------------------------
__global__ void gcgru_proj(const float* __restrict__ h, const float* __restrict__ pw,
                           const float* __restrict__ pb, float* __restrict__ out,
                           float* __restrict__ ybuf, int t) {
    int idx = blockIdx.x * blockDim.x + threadIdx.x;    // b*1024 + i
    if (idx >= ROWS) return;
    const float* hr = h + (size_t)idx * HH;
    float acc = pb[0];
    #pragma unroll
    for (int o = 0; o < HH; ++o) acc += hr[o] * pw[o];
    int b = idx >> 10, i = idx & 1023;
    out[((size_t)(b * 12 + t)) * NN + i] = acc;
    ybuf[idx] = acc;
}

// ------------------------------- host --------------------------------------
extern "C" void kernel_launch(void* const* d_in, const int* in_sizes, int n_in,
                              void* d_out, int out_size, void* d_ws, size_t ws_size,
                              hipStream_t stream) {
    const float* x      = (const float*)d_in[0];
    const float* G      = (const float*)d_in[1];
    const float* enc_Wg = (const float*)d_in[2];
    const float* enc_bg = (const float*)d_in[3];
    const float* enc_Wu = (const float*)d_in[4];
    const float* enc_bu = (const float*)d_in[5];
    const float* dec_Wg = (const float*)d_in[6];
    const float* dec_bg = (const float*)d_in[7];
    const float* dec_Wu = (const float*)d_in[8];
    const float* dec_bu = (const float*)d_in[9];
    const float* proj_W = (const float*)d_in[10];
    const float* proj_b = (const float*)d_in[11];
    float* out = (float*)d_out;

    char* ws = (char*)d_ws;
    size_t off = 0;
    auto carve = [&](size_t bytes) -> void* {
        void* p = ws + off;
        off = (off + bytes + 255) & ~(size_t)255;
        return p;
    };
    unsigned short* Gb   = (unsigned short*)carve((size_t)2 * NN * NN * 2);
    unsigned short* XT   = (unsigned short*)carve((size_t)QTOT * NN * 2);
    unsigned short* A2   = (unsigned short*)carve((size_t)ROWS * KPAD * 2);
    float*          zr   = (float*)carve((size_t)ROWS * 128 * 4);
    float*          h    = (float*)carve((size_t)ROWS * HH * 4);
    float*          ybuf = (float*)carve((size_t)ROWS * 4);
    unsigned short* WgTe = (unsigned short*)carve((size_t)128 * KPAD * 2);
    unsigned short* WuTe = (unsigned short*)carve((size_t)64  * KPAD * 2);
    unsigned short* WgTd = (unsigned short*)carve((size_t)128 * KPAD * 2);
    unsigned short* WuTd = (unsigned short*)carve((size_t)64  * KPAD * 2);

    // ---- init (every call: deterministic) ----
    gcgru_zero_f32<<<(ROWS * HH + 255) / 256, 256, 0, stream>>>(h, ROWS * HH);
    gcgru_zero_f32<<<(ROWS + 255) / 256, 256, 0, stream>>>(ybuf, ROWS);
    gcgru_pad_a2<<<ROWS / 256, 256, 0, stream>>>(A2);
    gcgru_conv_g<<<(2 * NN * NN + 255) / 256, 256, 0, stream>>>(G, Gb);
    gcgru_prep_w<<<(128 * KPAD + 255) / 256, 256, 0, stream>>>(enc_Wg, 128, WgTe);
    gcgru_prep_w<<<(64  * KPAD + 255) / 256, 256, 0, stream>>>(enc_Wu, 64,  WuTe);
    gcgru_prep_w<<<(128 * KPAD + 255) / 256, 256, 0, stream>>>(dec_Wg, 128, WgTd);
    gcgru_prep_w<<<(64  * KPAD + 255) / 256, 256, 0, stream>>>(dec_Wu, 64,  WuTd);

    auto cell = [&](const float* xp, int xstride, const unsigned short* WgT,
                    const float* bg, const unsigned short* WuT, const float* bu) {
        // gate GCN: sup = [X, G1X, G2X] -> sigmoid -> z,r
        gcgru_build<<<QTOT, 256, 0, stream>>>(xp, xstride, h, zr, 0, XT, A2);
        gcgru_gemm_graph<<<dim3(QTOT / 64, NN / 128, 2), 128, 0, stream>>>(Gb, XT, A2);
        gcgru_gemm_fuse<<<dim3(2, ROWS / 64, 1), 128, 0, stream>>>(A2, WgT, bg, 0, zr, h);
        // candidate GCN on [x, r*h] -> tanh -> GRU update of h
        gcgru_build<<<QTOT, 256, 0, stream>>>(xp, xstride, h, zr, 1, XT, A2);
        gcgru_gemm_graph<<<dim3(QTOT / 64, NN / 128, 2), 128, 0, stream>>>(Gb, XT, A2);
        gcgru_gemm_fuse<<<dim3(1, ROWS / 64, 1), 128, 0, stream>>>(A2, WuT, bu, 1, zr, h);
    };

    for (int t = 0; t < 12; ++t)
        cell(x + (size_t)t * NN, 12 * NN, WgTe, enc_bg, WuTe, enc_bu);

    for (int t = 0; t < 12; ++t) {
        cell(ybuf, NN, WgTd, dec_bg, WuTd, dec_bu);
        gcgru_proj<<<(ROWS + 255) / 256, 256, 0, stream>>>(h, proj_W, proj_b, out, ybuf, t);
    }
}
